// SingleScale_DenseDynGCN_LSTM_75746043232976
// MI455X (gfx1250) — compile-verified, matrix operand order checked by reading, not run on hardware
//
#include <hip/hip_runtime.h>
#include <math.h>

// ---------------------------------------------------------------------------
// CDNA5 (gfx1250) wave32 WMMA types & fragment helpers
// ---------------------------------------------------------------------------
typedef __attribute__((ext_vector_type(16))) _Float16 v16h;
typedef __attribute__((ext_vector_type(8)))  float    v8f;

#define WMMA_F32_F16(a, b, c) \
  __builtin_amdgcn_wmma_f32_16x16x32_f16(false, (a), false, (b), (short)0, (c), false, false)

// A fragment: 16x32 (MxK) f16 from row-major LDS (ld = row stride in halfs).
// Lanes 0-15: M=lane, K = 0..7 then 16..23; lanes 16-31: M=lane-16, K = 8..15 then 24..31.
static __device__ inline v16h frag_a_lds(const _Float16* s, int m0, int k0, int ld) {
  const int l = threadIdx.x & 31;
  const _Float16* row = s + (size_t)(m0 + (l & 15)) * ld + k0 + ((l >> 4) << 3);
  v16h a;
#pragma unroll
  for (int i = 0; i < 8; ++i) { a[i] = row[i]; a[i + 8] = row[16 + i]; }
  return a;
}

// B fragment: 32x16 (KxN) from row-major KxN f16 source (ld = row stride).
static __device__ inline v16h frag_b_rm(const _Float16* s, int k0, int n0, int ld) {
  const int l = threadIdx.x & 31;
  const _Float16* col = s + (size_t)(k0 + ((l >> 4) << 3)) * ld + n0 + (l & 15);
  v16h b;
#pragma unroll
  for (int i = 0; i < 8; ++i) { b[i] = col[i * ld]; b[i + 8] = col[(16 + i) * ld]; }
  return b;
}

// B fragment where B[k][n] = W[n][k] : W row-major N x K f16 (LDS or global).
static __device__ inline v16h frag_b_wT(const _Float16* W, int n0, int k0, int ld) {
  const int l = threadIdx.x & 31;
  const _Float16* row = W + (size_t)(n0 + (l & 15)) * ld + k0 + ((l >> 4) << 3);
  v16h b;
#pragma unroll
  for (int i = 0; i < 8; ++i) { b[i] = row[i]; b[i + 8] = row[16 + i]; }
  return b;
}

static __device__ inline float sigf(float x) { return 1.0f / (1.0f + __expf(-x)); }

// CDNA5 async global->LDS copy (ASYNCcnt-tracked, no VGPR round trip).
// LDS address from the first VGPR operand; global address is SGPR base plus VGPR offset.
static __device__ inline void async_copy_b128(unsigned lds_off, const void* gbase,
                                              unsigned goff) {
  asm volatile("global_load_async_to_lds_b128 %0, %1, %2"
               :: "v"(lds_off), "v"(goff), "s"(gbase)
               : "memory");
}
static __device__ inline void async_wait0() {
  asm volatile("s_wait_asynccnt 0x0" ::: "memory");
}

#define Bz 32
#define Cz 19
#define Tz 512
#define Fz 128

// ---------------------------------------------------------------------------
// Kernel 0: one-shot fp32 -> fp16 weight image conversion.
// ---------------------------------------------------------------------------
__global__ __launch_bounds__(256) void f16cvt(const float* __restrict__ src,
                                              _Float16* __restrict__ dst, int n) {
  const int i = blockIdx.x * 256 + threadIdx.x;
  if (i < n) dst[i] = (_Float16)src[i];
}

// ---------------------------------------------------------------------------
// Kernel 1: dynamic-adjacency GCN per (b,t); one 256-thread workgroup each.
// ---------------------------------------------------------------------------
__global__ __launch_bounds__(256) void gcn_kernel(
    const float* __restrict__ x,
    const _Float16* __restrict__ w1h, const float* __restrict__ b1,
    const _Float16* __restrict__ w2h, const float* __restrict__ b2,
    const float* __restrict__ Wac, const float* __restrict__ bac,
    const float* __restrict__ Wfp, const float* __restrict__ bfp,
    float* __restrict__ seq /* [T*B][64] */) {
  __shared__ __align__(16) _Float16 wbuf[128 * 128];  // W1 f16, later W2 f16
  __shared__ __align__(16) _Float16 abuf[32 * 128];   // x f16 -> later g1 f16
  __shared__ __align__(16) _Float16 nbuf[32 * 128];   // xn f16 -> h1 f16 -> h2 f16 (ld 64)
  __shared__ float    adjf[32 * 32];
  __shared__ __align__(16) _Float16 adjh[32 * 32];
  __shared__ float    hdyn[32 * 64];
  __shared__ float    meanv[32], istdv[32], degv[32], logits[32];
  __shared__ float    pooled[64];
  __shared__ float    invsum;

  const int tid = threadIdx.x;
  const int w = tid >> 5;
  const int bt = blockIdx.x;
  const int batch = bt / Tz;
  const int tpos = bt % Tz;
  const float* xb = x + (size_t)batch * Cz * Tz * Fz + (size_t)tpos * Fz;

  // ---- kick off async copy of W1 f16 image (32 KB) into LDS
  {
    const unsigned wb = (unsigned)(size_t)(&wbuf[0]);
#pragma unroll
    for (int i = 0; i < 8; ++i) {
      const unsigned off = (unsigned)(i * 256 + tid) * 16u;
      async_copy_b128(wb + off, w1h, off);
    }
  }

  // ---- load x tile as f16, padded to 32 rows
  for (int idx = tid; idx < 32 * 128; idx += 256) {
    const int c = idx >> 7, f = idx & 127;
    const float v = (c < Cz) ? xb[(size_t)c * (Tz * Fz) + f] : 0.0f;
    abuf[idx] = (_Float16)v;
  }
  __syncthreads();

  // ---- per-channel mean / inv-std
  if (tid < Cz) {
    float s = 0.0f;
    for (int f = 0; f < Fz; ++f) s += (float)abuf[tid * 128 + f];
    const float mean = s * (1.0f / Fz);
    float q = 0.0f;
    for (int f = 0; f < Fz; ++f) { const float d = (float)abuf[tid * 128 + f] - mean; q += d * d; }
    const float sd = fmaxf(sqrtf(q * (1.0f / Fz)), 1e-6f);
    meanv[tid] = mean;
    istdv[tid] = 1.0f / sd;
  }
  __syncthreads();
  for (int idx = tid; idx < 32 * 128; idx += 256) {
    const int c = idx >> 7;
    const float v = (c < Cz) ? ((float)abuf[idx] - meanv[c]) * istdv[c] : 0.0f;
    nbuf[idx] = (_Float16)v;
  }
  __syncthreads();

  // ---- corr = xn @ xn^T / F  (M=32,N=32,K=128) -> |corr| into adjf
  if (w < 4) {
    const int mi = w >> 1, ni = w & 1;
    v8f acc = {};
#pragma unroll
    for (int kt = 0; kt < 4; ++kt) {
      const v16h a = frag_a_lds(nbuf, mi * 16, kt * 32, 128);
      const v16h bm = frag_b_wT(nbuf, ni * 16, kt * 32, 128);
      acc = WMMA_F32_F16(a, bm, acc);
    }
    const int l = tid & 31, col = ni * 16 + (l & 15), rb = mi * 16 + ((l >> 4) << 3);
#pragma unroll
    for (int i = 0; i < 8; ++i)
      adjf[(rb + i) * 32 + col] = fabsf(acc[i] * (1.0f / Fz));
  }
  __syncthreads();

  // ---- threshold + eps*I
  for (int idx = tid; idx < 1024; idx += 256) {
    const int c = idx >> 5, d = idx & 31;
    float m = 0.0f;
    if (c < Cz && d < Cz) {
      const float a = adjf[idx];
      m = (a > 0.8f) ? a : 0.0f;
      if (c == d) m += 1e-6f;
    }
    adjf[idx] = m;
  }
  __syncthreads();
  if (tid < Cz) {
    float s = 0.0f;
    for (int d = 0; d < Cz; ++d) s += adjf[tid * 32 + d];
    degv[tid] = fmaxf(rsqrtf(s), 1e-6f);
  }
  __syncthreads();
  for (int idx = tid; idx < 1024; idx += 256) {
    const int c = idx >> 5, d = idx & 31;
    const float v = (c < Cz && d < Cz) ? degv[c] * degv[d] * adjf[idx] : 0.0f;
    adjf[idx] = v;
    adjh[idx] = (_Float16)v;
  }
  async_wait0();     // W1 image resident in LDS
  __syncthreads();

  // ---- h1 = relu(x @ W1^T + b1)  (M=32,N=128,K=128) -> nbuf f16
#pragma unroll
  for (int r = 0; r < 2; ++r) {
    const int tt = w * 2 + r;
    const int mi = tt >> 3, ni = tt & 7;
    v8f acc = {};
#pragma unroll
    for (int kt = 0; kt < 4; ++kt) {
      const v16h a = frag_a_lds(abuf, mi * 16, kt * 32, 128);
      const v16h bm = frag_b_wT(wbuf, ni * 16, kt * 32, 128);
      acc = WMMA_F32_F16(a, bm, acc);
    }
    const int l = tid & 31, col = ni * 16 + (l & 15), rb = mi * 16 + ((l >> 4) << 3);
    const float bias = b1[col];
#pragma unroll
    for (int i = 0; i < 8; ++i)
      nbuf[(rb + i) * 128 + col] = (_Float16)fmaxf(acc[i] + bias, 0.0f);
  }
  __syncthreads();

  // ---- g1 = relu(adj @ h1)  (M=32,N=128,K=32) -> abuf f16
#pragma unroll
  for (int r = 0; r < 2; ++r) {
    const int tt = w * 2 + r;
    const int mi = tt >> 3, ni = tt & 7;
    const v16h a = frag_a_lds(adjh, mi * 16, 0, 32);
    const v16h bm = frag_b_rm(nbuf, 0, ni * 16, 128);
    v8f acc = {};
    acc = WMMA_F32_F16(a, bm, acc);
    const int l = tid & 31, col = ni * 16 + (l & 15), rb = mi * 16 + ((l >> 4) << 3);
#pragma unroll
    for (int i = 0; i < 8; ++i)
      abuf[(rb + i) * 128 + col] = (_Float16)fmaxf(acc[i], 0.0f);
  }
  __syncthreads();

  // ---- async copy W2 f16 image (16 KB) into wbuf (all W1 readers are done)
  {
    const unsigned wb = (unsigned)(size_t)(&wbuf[0]);
#pragma unroll
    for (int i = 0; i < 4; ++i) {
      const unsigned off = (unsigned)(i * 256 + tid) * 16u;
      async_copy_b128(wb + off, w2h, off);
    }
    async_wait0();
  }
  __syncthreads();

  // ---- h2 = g1 @ W2^T + b2  (M=32,N=64,K=128) -> nbuf f16 (ld 64)
  {
    const int mi = w >> 2, ni = w & 3;
    v8f acc = {};
#pragma unroll
    for (int kt = 0; kt < 4; ++kt) {
      const v16h a = frag_a_lds(abuf, mi * 16, kt * 32, 128);
      const v16h bm = frag_b_wT(wbuf, ni * 16, kt * 32, 128);
      acc = WMMA_F32_F16(a, bm, acc);
    }
    const int l = tid & 31, col = ni * 16 + (l & 15), rb = mi * 16 + ((l >> 4) << 3);
    const float bias = b2[col];
#pragma unroll
    for (int i = 0; i < 8; ++i)
      nbuf[(rb + i) * 64 + col] = (_Float16)(acc[i] + bias);
  }
  __syncthreads();

  // ---- h_dyn = relu(adj @ h2)  (M=32,N=64,K=32) -> hdyn f32
  {
    const int mi = w >> 2, ni = w & 3;
    const v16h a = frag_a_lds(adjh, mi * 16, 0, 32);
    const v16h bm = frag_b_rm(nbuf, 0, ni * 16, 64);
    v8f acc = {};
    acc = WMMA_F32_F16(a, bm, acc);
    const int l = tid & 31, col = ni * 16 + (l & 15), rb = mi * 16 + ((l >> 4) << 3);
#pragma unroll
    for (int i = 0; i < 8; ++i)
      hdyn[(rb + i) * 64 + col] = fmaxf(acc[i], 0.0f);
  }
  __syncthreads();

  // ---- channel attention: h_cat = [h_dyn, 0] so only first 64 features matter
  if (tid < Cz) {
    float s = bac[0];
    for (int j = 0; j < 64; ++j) s += hdyn[tid * 64 + j] * Wac[j];
    logits[tid] = s;
  }
  __syncthreads();
  if (tid == 0) {
    float mx = -1e30f;
    for (int c = 0; c < Cz; ++c) mx = fmaxf(mx, logits[c]);
    float sum = 0.0f;
    for (int c = 0; c < Cz; ++c) { const float e = __expf(logits[c] - mx); logits[c] = e; sum += e; }
    invsum = 1.0f / sum;
  }
  __syncthreads();
  if (tid < 64) {
    float s = 0.0f;
    for (int c = 0; c < Cz; ++c) s += logits[c] * hdyn[c * 64 + tid];
    pooled[tid] = s * invsum;
  }
  __syncthreads();
  // ---- front projection: seq = pooled @ Wfp^T + bfp (pooled[64:]==0)
  if (tid < 64) {
    float s = bfp[tid];
    for (int j = 0; j < 64; ++j) s += pooled[j] * Wfp[tid * 128 + j];
    seq[((size_t)tpos * Bz + batch) * 64 + tid] = s;
  }
}

// ---------------------------------------------------------------------------
// Kernel 2: hoisted LSTM input projection G = X @ Wih^T + (bih + bhh)
// X: [T*B][in_dim] f32, Wih f16, G: [T*B][256] f32.  32 rows per block.
// ---------------------------------------------------------------------------
__global__ __launch_bounds__(256) void lstm_in_gemm(
    const float* __restrict__ X, int in_dim,
    const _Float16* __restrict__ Wih, const float* __restrict__ bih,
    const float* __restrict__ bhh, float* __restrict__ G) {
  __shared__ __align__(16) _Float16 xh[32 * 128];
  const int tid = threadIdx.x, w = tid >> 5;
  const size_t r0 = (size_t)blockIdx.x * 32;

  __builtin_prefetch(Wih + tid * 16, 0, 1);

  for (int idx = tid; idx < 32 * 128; idx += 256) {
    const int rr = idx >> 7, f = idx & 127;
    xh[idx] = (_Float16)((f < in_dim) ? X[(r0 + rr) * in_dim + f] : 0.0f);
  }
  __syncthreads();

  const int ksteps = in_dim >> 5;
#pragma unroll
  for (int r = 0; r < 4; ++r) {
    const int tt = w * 4 + r;  // 0..31 : 2 M-tiles x 16 N-tiles
    const int mi = tt >> 4, ni = tt & 15;
    v8f acc = {};
    for (int kt = 0; kt < ksteps; ++kt) {
      const v16h a = frag_a_lds(xh, mi * 16, kt * 32, 128);
      const v16h bm = frag_b_wT(Wih, ni * 16, kt * 32, in_dim);
      acc = WMMA_F32_F16(a, bm, acc);
    }
    const int l = tid & 31, col = ni * 16 + (l & 15), rb = mi * 16 + ((l >> 4) << 3);
    const float bias = bih[col] + bhh[col];
#pragma unroll
    for (int i = 0; i < 8; ++i)
      G[(r0 + rb + i) * 256 + col] = acc[i] + bias;
  }
}

// ---------------------------------------------------------------------------
// Kernel 3: LSTM recurrence. blockIdx.x = direction (0 fwd, 1 bwd).
// Per step: g = G[t] + h @ Whh^T (WMMA, M=32=B, N=256, K=64), cell update.
// Whh B-fragments live in registers for the whole sequence.
// ---------------------------------------------------------------------------
__global__ __launch_bounds__(256) void lstm_recurrent(
    const float* __restrict__ Gf, const float* __restrict__ Gb,
    const _Float16* __restrict__ WhhF, const _Float16* __restrict__ WhhB,
    float* __restrict__ HS /* [T*B][128] */) {
  __shared__ __align__(16) _Float16 hbuf[32 * 64];
  __shared__ float    cbuf[32 * 64];
  __shared__ float    gbuf[32 * 256];

  const int tid = threadIdx.x, w = tid >> 5;
  const int dir = blockIdx.x;
  const float* G = dir ? Gb : Gf;
  const _Float16* Whh = dir ? WhhB : WhhF;
  const int coff = dir * 64;

  for (int idx = tid; idx < 32 * 64; idx += 256) { hbuf[idx] = (_Float16)0.0f; cbuf[idx] = 0.0f; }

  // Preload Whh^T fragments: wave w owns N-tiles {w, w+8}
  v16h Bf[2][2];
#pragma unroll
  for (int r = 0; r < 2; ++r) {
    const int ni = w + r * 8;
#pragma unroll
    for (int kt = 0; kt < 2; ++kt)
      Bf[r][kt] = frag_b_wT(Whh, ni * 16, kt * 32, 64);
  }
  __syncthreads();

  const int l = tid & 31, lcol = l & 15, lrb = (l >> 4) << 3;
  for (int s = 0; s < Tz; ++s) {
    const int t = dir ? (Tz - 1 - s) : s;
    const float* Gt = G + (size_t)t * Bz * 256;
    if (s + 1 < Tz) {  // hide next step's 32KB gate fetch behind this step
      const int tn = dir ? (t - 1) : (t + 1);
      __builtin_prefetch(G + (size_t)tn * Bz * 256 + tid * 32, 0, 1);
    }
#pragma unroll
    for (int r = 0; r < 2; ++r) {
      const int ni = w + r * 8;
#pragma unroll
      for (int mi = 0; mi < 2; ++mi) {
        const int col = ni * 16 + lcol, rowb = mi * 16 + lrb;
        v8f acc;
#pragma unroll
        for (int i = 0; i < 8; ++i) acc[i] = Gt[(rowb + i) * 256 + col];
#pragma unroll
        for (int kt = 0; kt < 2; ++kt) {
          const v16h a = frag_a_lds(hbuf, mi * 16, kt * 32, 64);
          acc = WMMA_F32_F16(a, Bf[r][kt], acc);
        }
#pragma unroll
        for (int i = 0; i < 8; ++i) gbuf[(rowb + i) * 256 + col] = acc[i];
      }
    }
    __syncthreads();
    // cell update: 32 batches x 64 hidden = 2048 cells / 256 threads
#pragma unroll
    for (int e = 0; e < 8; ++e) {
      const int id = e * 256 + tid;
      const int bb = id >> 6, j = id & 63;
      const float gi = gbuf[bb * 256 + j];
      const float gf = gbuf[bb * 256 + 64 + j];
      const float gg = gbuf[bb * 256 + 128 + j];
      const float go = gbuf[bb * 256 + 192 + j];
      const float c = sigf(gf) * cbuf[bb * 64 + j] + sigf(gi) * tanhf(gg);
      const float h = sigf(go) * tanhf(c);
      cbuf[bb * 64 + j] = c;
      hbuf[bb * 64 + j] = (_Float16)h;
      HS[((size_t)t * Bz + bb) * 128 + coff + j] = h;
    }
    __syncthreads();
  }
}

// ---------------------------------------------------------------------------
// Kernel 4: temporal attention pooling + final projection. One block per batch.
// ---------------------------------------------------------------------------
__global__ __launch_bounds__(256) void attn_proj(
    const float* __restrict__ HS, const float* __restrict__ Wat,
    const float* __restrict__ bat, const float* __restrict__ Wp,
    const float* __restrict__ bp, float* __restrict__ out) {
  __shared__ float logit[Tz];
  __shared__ float ctx[128];
  __shared__ float invsum;
  const int tid = threadIdx.x, b = blockIdx.x;

  for (int t = tid; t < Tz; t += 256) {
    const float* h = HS + ((size_t)t * Bz + b) * 128;
    float s = bat[0];
    for (int j = 0; j < 128; ++j) s += h[j] * Wat[j];
    logit[t] = s;
  }
  __syncthreads();
  if (tid == 0) {
    float mx = -1e30f;
    for (int t = 0; t < Tz; ++t) mx = fmaxf(mx, logit[t]);
    float sum = 0.0f;
    for (int t = 0; t < Tz; ++t) { const float e = __expf(logit[t] - mx); logit[t] = e; sum += e; }
    invsum = 1.0f / sum;
  }
  __syncthreads();
  if (tid < 128) {
    float s = 0.0f;
    for (int t = 0; t < Tz; ++t) s += logit[t] * HS[((size_t)t * Bz + b) * 128 + tid];
    ctx[tid] = s * invsum;
  }
  __syncthreads();
  if (tid < 128) {
    float s = bp[tid];
    for (int j = 0; j < 128; ++j) s += ctx[j] * Wp[tid * 128 + j];
    out[b * 128 + tid] = s;
  }
}

// ---------------------------------------------------------------------------
extern "C" void kernel_launch(void* const* d_in, const int* in_sizes, int n_in,
                              void* d_out, int out_size, void* d_ws, size_t ws_size,
                              hipStream_t stream) {
  const float* x   = (const float*)d_in[0];
  const float* W1  = (const float*)d_in[1];
  const float* b1  = (const float*)d_in[2];
  const float* W2  = (const float*)d_in[3];
  const float* b2  = (const float*)d_in[4];
  const float* Wac = (const float*)d_in[5];
  const float* bac = (const float*)d_in[6];
  const float* Wfp = (const float*)d_in[7];
  const float* bfp = (const float*)d_in[8];
  // lstm_params: 8 x (Wih, Whh, bih, bhh) flattened, order: (l0,fwd),(l0,bwd),(l1,fwd),(l1,bwd)
  auto lp = [&](int layer, int dir, int k) -> const float* {
    return (const float*)d_in[9 + 4 * (2 * layer + dir) + k];
  };
  const float* Wat = (const float*)d_in[41];
  const float* bat = (const float*)d_in[42];
  const float* Wp  = (const float*)d_in[43];
  const float* bp  = (const float*)d_in[44];
  float* outp = (float*)d_out;

  // workspace layout: fp32 activations, then fp16 weight images
  float* seqw = (float*)d_ws;                          // [T*B][64]
  float* hs0  = seqw + (size_t)Tz * Bz * 64;           // [T*B][128]
  float* hs1  = hs0 + (size_t)Tz * Bz * 128;           // [T*B][128]
  float* Gf   = hs1 + (size_t)Tz * Bz * 128;           // [T*B][256]
  float* Gb   = Gf + (size_t)Tz * Bz * 256;            // [T*B][256]
  _Float16* h16 = (_Float16*)(Gb + (size_t)Tz * Bz * 256);
  _Float16* w1h = h16;                 h16 += 128 * 128;
  _Float16* w2h = h16;                 h16 += 64 * 128;
  _Float16* wih[4];                    // per (layer,dir)
  _Float16* whh[4];
  const int ihn[4] = {256 * 64, 256 * 64, 256 * 128, 256 * 128};
  for (int i = 0; i < 4; ++i) { wih[i] = h16; h16 += ihn[i]; }
  for (int i = 0; i < 4; ++i) { whh[i] = h16; h16 += 256 * 64; }

  // ---- one-shot weight conversions (tiny)
  f16cvt<<<(128 * 128 + 255) / 256, 256, 0, stream>>>(W1, w1h, 128 * 128);
  f16cvt<<<(64 * 128 + 255) / 256, 256, 0, stream>>>(W2, w2h, 64 * 128);
  for (int i = 0; i < 4; ++i) {
    const int layer = i >> 1, dir = i & 1;
    f16cvt<<<(ihn[i] + 255) / 256, 256, 0, stream>>>(lp(layer, dir, 0), wih[i], ihn[i]);
    f16cvt<<<(256 * 64 + 255) / 256, 256, 0, stream>>>(lp(layer, dir, 1), whh[i], 256 * 64);
  }

  // ---- GCN + channel attention + front projection over all 16384 (b,t) tiles
  gcn_kernel<<<Bz * Tz, 256, 0, stream>>>(x, w1h, b1, w2h, b2, Wac, bac, Wfp, bfp, seqw);

  // ---- BiLSTM layer 0 (in_dim = 64)
  lstm_in_gemm<<<(Tz * Bz) / 32, 256, 0, stream>>>(seqw, 64, wih[0], lp(0, 0, 2), lp(0, 0, 3), Gf);
  lstm_in_gemm<<<(Tz * Bz) / 32, 256, 0, stream>>>(seqw, 64, wih[1], lp(0, 1, 2), lp(0, 1, 3), Gb);
  lstm_recurrent<<<2, 256, 0, stream>>>(Gf, Gb, whh[0], whh[1], hs0);

  // ---- BiLSTM layer 1 (in_dim = 128)
  lstm_in_gemm<<<(Tz * Bz) / 32, 256, 0, stream>>>(hs0, 128, wih[2], lp(1, 0, 2), lp(1, 0, 3), Gf);
  lstm_in_gemm<<<(Tz * Bz) / 32, 256, 0, stream>>>(hs0, 128, wih[3], lp(1, 1, 2), lp(1, 1, 3), Gb);
  lstm_recurrent<<<2, 256, 0, stream>>>(Gf, Gb, whh[2], whh[3], hs1);

  // ---- temporal attention + final projection
  attn_proj<<<Bz, 256, 0, stream>>>(hs1, Wat, bat, Wp, bp, outp);

  (void)in_sizes; (void)n_in; (void)out_size; (void)ws_size;
}
